// PreyCritic_13589276525228
// MI455X (gfx1250) — matrix operand order repaired and checked by reading, not compiled, not run on hardware
//
#include <hip/hip_runtime.h>
#include <hip/hip_bf16.h>

typedef _Float16 half_t;
typedef __attribute__((ext_vector_type(16))) _Float16 v16h;
typedef __attribute__((ext_vector_type(8)))  float    v8f;

#define BH   128   // hidden dim H
#define NPRD 8
#define NPRY 8
#define NOBS 16
#define NE   32    // total entities

// ---------------------------------------------------------------------------
// Prep: convert six HxH f32 weight matrices to f16, transposed (WT[n][k] = W[k][n])
// so WMMA B-operand fetches are two contiguous b128 loads per lane.
// ws16 layout: [0]=WxT1 [1]=WgT1 [2]=WxT2 [3]=WgT2 [4]=fW1T [5]=fW2T, 16384 halves each.
// ---------------------------------------------------------------------------
__global__ __launch_bounds__(256) void prep_weights(
    const float* __restrict__ c1_Wl, const float* __restrict__ c1_Wg,
    const float* __restrict__ c2_Wl, const float* __restrict__ c2_Wg,
    const float* __restrict__ f_W1,  const float* __restrict__ f_W2,
    half_t* __restrict__ ws16)
{
  int i = blockIdx.x * blockDim.x + threadIdx.x;   // 6*16384 threads
  if (i >= 6 * 16384) return;
  int m   = i >> 14;
  int idx = i & 16383;
  int n   = idx >> 7;          // output row (= original column)
  int k   = idx & 127;         // output col (= original row)
  const float* src;
  switch (m) {
    case 0: src = c1_Wl; break;   // Wx1 = first 128 rows of (130,128)
    case 1: src = c1_Wg; break;
    case 2: src = c2_Wl; break;
    case 3: src = c2_Wg; break;
    case 4: src = f_W1;  break;
    default: src = f_W2; break;
  }
  ws16[m * 16384 + n * 128 + k] = (half_t)src[k * 128 + n];
}

// ---------------------------------------------------------------------------
// WMMA operand loaders (wave32; A and B both use contiguous k order k = 16*hi + s)
// ---------------------------------------------------------------------------
__device__ __forceinline__ v16h load_A_lds(const half_t* Abase, int mtile, int kstep, int lane) {
  // A is 16x32 f16: lane(m=lane%16, hi=lane/16); halves[s]:
  //   K = 16*hi + s (contiguous, matching B)
  int m  = mtile * 16 + (lane & 15);
  int hi = lane >> 4;
  const half_t* r = Abase + m * BH + kstep * 32;
  union { v16h v; half_t h[16]; } A;
#pragma unroll
  for (int t = 0; t < 8; ++t) {
    A.h[t]     = r[16 * hi + t];
    A.h[8 + t] = r[16 * hi + 8 + t];
  }
  return A.v;
}

__device__ __forceinline__ v16h load_B_ws(const half_t* WT, int ntile, int kstep, int lane) {
  // B is 32x16 f16: lane(n=lane%16, hi=lane/16) holds K = 16*hi .. 16*hi+15 of column n.
  // WT is transposed (N-major), so these 16 halves are contiguous.
  int n  = ntile * 16 + (lane & 15);
  int hi = lane >> 4;
  const half_t* p = WT + n * BH + kstep * 32 + hi * 16;
  union { v16h v; half_t h[16]; } B;
#pragma unroll
  for (int t = 0; t < 16; ++t) B.h[t] = p[t];
  return B.v;
}

__device__ __forceinline__ v8f wmma_tile(const half_t* A_lds, const half_t* WT,
                                         int mtile, int ntile, int lane) {
  v8f acc = {};
#pragma unroll
  for (int ks = 0; ks < 4; ++ks) {   // K = 128 = 4 * 32
    v16h a = load_A_lds(A_lds, mtile, ks, lane);
    v16h b = load_B_ws(WT, ntile, ks, lane);
    acc = __builtin_amdgcn_wmma_f32_16x16x32_f16(false, a, false, b,
                                                 (short)0, acc, false, false);
  }
  return acc;
}

// ---------------------------------------------------------------------------
// Main kernel: one workgroup (128 threads = 4 waves) per batch element.
// ---------------------------------------------------------------------------
__global__ __launch_bounds__(128) void prey_critic_main(
    const float* __restrict__ pred,  const float* __restrict__ prey,
    const float* __restrict__ obst,  const unsigned char* __restrict__ alive_in,
    const float* __restrict__ action,
    const float* __restrict__ emb,   const float* __restrict__ W_obst,
    const float* __restrict__ b_obst,
    const float* __restrict__ W_act, const float* __restrict__ b_act,
    const float* __restrict__ W_pos, const float* __restrict__ b_pos,
    const float* __restrict__ c1_Wl, const float* __restrict__ c1_bl,
    const float* __restrict__ c1_bg,
    const float* __restrict__ c2_Wl, const float* __restrict__ c2_bl,
    const float* __restrict__ c2_bg,
    const float* __restrict__ f_b1,  const float* __restrict__ f_b2,
    const float* __restrict__ f_W3,  const float* __restrict__ f_b3,
    const half_t* __restrict__ ws16,
    float* __restrict__ out)
{
  __shared__ half_t X16[NE][BH];   // f16 GEMM A operand (x_all / x_all2 / MLP in)
  __shared__ half_t S16[NE][BH];   // f16 GEMM A operand (relu-sum result / h1)
  __shared__ float  Uf[NE][BH];    // u[j][h] f32 (also O2 / h2 staging)
  __shared__ float  Pw[NE][BH];    // posW[i][h] f32 (also final-reduce scratch)
  __shared__ float  Xa[NPRY][BH];  // x_action f32
  __shared__ float  posf[NE][2];
  __shared__ float  aliveS[NPRY];
  __shared__ float  deadOff[NE];   // -1e31 for dead-prey source rows, else 0
  __shared__ float  scaleR[NE];    // alive for prey rows, else 1
  __shared__ float  preyM[NE];     // 1 for prey rows, else 0

  const int b    = blockIdx.x;
  const int tid  = threadIdx.x;
  const int wave = tid >> 5;
  const int lane = tid & 31;
  const int nl   = lane & 15;      // WMMA column within tile
  const int hi   = lane >> 4;

  const half_t* WxT1 = ws16;
  const half_t* WgT1 = ws16 + 16384;
  const half_t* WxT2 = ws16 + 2 * 16384;
  const half_t* WgT2 = ws16 + 3 * 16384;
  const half_t* fW1T = ws16 + 4 * 16384;
  const half_t* fW2T = ws16 + 5 * 16384;

  // ---- stage positions + alive + per-row mask constants -------------------
  if (tid < NPRD * 2) {
    posf[tid >> 1][tid & 1] = pred[b * NPRD * 2 + tid];
  } else if (tid < (NPRD + NPRY) * 2) {
    int t = tid - NPRD * 2;
    posf[NPRD + (t >> 1)][t & 1] = prey[b * NPRY * 2 + t];
  } else if (tid < (NPRD + NPRY + NOBS) * 2) {
    int t = tid - (NPRD + NPRY) * 2;
    posf[NPRD + NPRY + (t >> 1)][t & 1] = obst[b * NOBS * 3 + (t >> 1) * 3 + (t & 1)];
  }
  if (tid < NPRY) aliveS[tid] = alive_in[b * NPRY + tid] ? 1.0f : 0.0f;
  if (tid >= 64 && tid < 64 + NE) {
    int m = tid - 64;
    int isPrey = (m >= NPRD) && (m < NPRD + NPRY);
    float al = isPrey ? (alive_in[b * NPRY + (m - NPRD)] ? 1.0f : 0.0f) : 1.0f;
    preyM[m]   = isPrey ? 1.0f : 0.0f;
    scaleR[m]  = al;
    deadOff[m] = (isPrey && al == 0.0f) ? -1e31f : 0.0f;
  }
  __syncthreads();

  // ---- x_action -----------------------------------------------------------
  for (int idx = tid; idx < NPRY * BH; idx += 128) {
    int y = idx >> 7, h = idx & 127;
    float a = action[b * NPRY + y] * 3.14159265358979f;
    float s, c;
    __sincosf(a, &s, &c);
    Xa[y][h] = s * W_act[h] + c * W_act[BH + h] + b_act[h];
  }
  __syncthreads();

  // ---- x_all (f16) + posW1 -----------------------------------------------
  const float* Wp1 = c1_Wl + 128 * BH;   // rows 128,129 of (130,128)
  for (int idx = tid; idx < NE * BH; idx += 128) {
    int e = idx >> 7, h = idx & 127;
    float v;
    if (e < NPRD)             v = emb[h];
    else if (e < NPRD + NPRY) { int y = e - NPRD; v = (emb[BH + h] + Xa[y][h]) * aliveS[y]; }
    else { int o = e - NPRD - NPRY;
           v = emb[2 * BH + h] + obst[b * NOBS * 3 + o * 3 + 2] * W_obst[h] + b_obst[h]; }
    X16[e][h] = (half_t)v;
    Pw[e][h]  = posf[e][0] * Wp1[h] + posf[e][1] * Wp1[BH + h];
  }
  __syncthreads();

  // ======================= conv1: u = X @ Wx1 + posW + bl (branchless mask)
  for (int tt = 0; tt < 4; ++tt) {
    int nt = wave * 2 + (tt & 1), mt = tt >> 1;
    int c  = nt * 16 + nl;
    v8f acc = wmma_tile(&X16[0][0], WxT1, mt, nt, lane);
    float blc = c1_bl[c];
#pragma unroll
    for (int r = 0; r < 8; ++r) {
      int m = mt * 16 + r + 8 * hi;
      Uf[m][c] = acc[r] + Pw[m][c] + blc + deadOff[m];
    }
  }
  __syncthreads();

  // ---- S[i][h] = sum_j relu(u[j][h] - posW[i][h]) ------------------------
  {
    int h = tid;                       // 128 threads <-> 128 channels
    float u[NE];
#pragma unroll
    for (int j = 0; j < NE; ++j) u[j] = Uf[j][h];
    for (int i = 0; i < NE; ++i) {
      float p = Pw[i][h], acc = 0.f;
#pragma unroll
      for (int j = 0; j < NE; ++j) acc += fmaxf(u[j] - p, 0.f);
      S16[i][h] = (half_t)acc;
    }
  }
  __syncthreads();

  // ---- O1 = S @ Wg1 + bg ; build x_all2 (f16), branchless residual --------
  for (int tt = 0; tt < 4; ++tt) {
    int nt = wave * 2 + (tt & 1), mt = tt >> 1;
    int c  = nt * 16 + nl;
    v8f acc = wmma_tile(&S16[0][0], WgT1, mt, nt, lane);
    float bgc = c1_bg[c];
#pragma unroll
    for (int r = 0; r < 8; ++r) {
      int m  = mt * 16 + r + 8 * hi;
      int yr = (m - NPRD) & (NPRY - 1);          // clamped prey index
      float v = (acc[r] + bgc) * scaleR[m] + Xa[yr][c] * preyM[m];
      X16[m][c] = (half_t)v;
    }
  }
  __syncthreads();

  // ======================= conv2 ==========================================
  const float* Wp2 = c2_Wl + 128 * BH;
  for (int idx = tid; idx < NE * BH; idx += 128) {
    int e = idx >> 7, h = idx & 127;
    Pw[e][h] = posf[e][0] * Wp2[h] + posf[e][1] * Wp2[BH + h];
  }
  __syncthreads();

  for (int tt = 0; tt < 4; ++tt) {
    int nt = wave * 2 + (tt & 1), mt = tt >> 1;
    int c  = nt * 16 + nl;
    v8f acc = wmma_tile(&X16[0][0], WxT2, mt, nt, lane);
    float blc = c2_bl[c];
#pragma unroll
    for (int r = 0; r < 8; ++r) {
      int m = mt * 16 + r + 8 * hi;
      Uf[m][c] = acc[r] + Pw[m][c] + blc + deadOff[m];
    }
  }
  __syncthreads();

  {
    int h = tid;
    float u[NE];
#pragma unroll
    for (int j = 0; j < NE; ++j) u[j] = Uf[j][h];
    for (int i = 0; i < NE; ++i) {
      float p = Pw[i][h], acc = 0.f;
#pragma unroll
      for (int j = 0; j < NE; ++j) acc += fmaxf(u[j] - p, 0.f);
      S16[i][h] = (half_t)acc;
    }
  }
  __syncthreads();

  // ---- O2 = S @ Wg2 + bg (f32 into Uf) ------------------------------------
  for (int tt = 0; tt < 4; ++tt) {
    int nt = wave * 2 + (tt & 1), mt = tt >> 1;
    int c  = nt * 16 + nl;
    v8f acc = wmma_tile(&S16[0][0], WgT2, mt, nt, lane);
    float bgc = c2_bg[c];
#pragma unroll
    for (int r = 0; r < 8; ++r) {
      int m = mt * 16 + r + 8 * hi;
      Uf[m][c] = acc[r] + bgc;
    }
  }
  __syncthreads();

  // ---- MLP input: rows 0..7 = prey features, 8..15 = zero pad -------------
  for (int idx = tid; idx < 16 * BH; idx += 128) {
    int r = idx >> 7, h = idx & 127;
    float v = 0.f;
    if (r < NPRY) {
      v = (Uf[NPRD + r][h] + Xa[r][h]) * aliveS[r]
        + posf[NPRD + r][0] * W_pos[h] + posf[NPRD + r][1] * W_pos[BH + h] + b_pos[h];
    }
    X16[r][h] = (half_t)v;
  }
  __syncthreads();

  // ---- h1 = relu(Fin @ fW1 + b1) ------------------------------------------
  for (int tt = 0; tt < 2; ++tt) {
    int nt = wave * 2 + tt;
    int c  = nt * 16 + nl;
    v8f acc = wmma_tile(&X16[0][0], fW1T, 0, nt, lane);
    float b1c = f_b1[c];
#pragma unroll
    for (int r = 0; r < 8; ++r) {
      int m = r + 8 * hi;
      S16[m][c] = (half_t)fmaxf(acc[r] + b1c, 0.f);
    }
  }
  __syncthreads();

  // ---- h2 = relu(h1 @ fW2 + b2) (f32 into Uf) -----------------------------
  for (int tt = 0; tt < 2; ++tt) {
    int nt = wave * 2 + tt;
    int c  = nt * 16 + nl;
    v8f acc = wmma_tile(&S16[0][0], fW2T, 0, nt, lane);
    float b2c = f_b2[c];
#pragma unroll
    for (int r = 0; r < 8; ++r) {
      int m = r + 8 * hi;
      Uf[m][c] = fmaxf(acc[r] + b2c, 0.f);
    }
  }
  __syncthreads();

  // ---- y = h2 @ f_W3 + b3 (parallel partial sums, reduce in LDS) ----------
  {
    int y = tid >> 4, seg = tid & 15;
    float acc = 0.f;
#pragma unroll
    for (int t = 0; t < 8; ++t) {
      int h = seg * 8 + t;
      acc += Uf[y][h] * f_W3[h];
    }
    Pw[y][seg] = acc;
  }
  __syncthreads();
  if (tid < NPRY) {
    float acc = f_b3[0];
#pragma unroll
    for (int s = 0; s < 16; ++s) acc += Pw[tid][s];
    out[b * NPRY + tid] = acc;
  }
}

// ---------------------------------------------------------------------------
extern "C" void kernel_launch(void* const* d_in, const int* in_sizes, int n_in,
                              void* d_out, int out_size, void* d_ws, size_t ws_size,
                              hipStream_t stream) {
  (void)n_in; (void)out_size; (void)ws_size;
  const float*         pred    = (const float*)d_in[0];
  const float*         prey    = (const float*)d_in[1];
  const float*         obst    = (const float*)d_in[2];
  const unsigned char* alive   = (const unsigned char*)d_in[3];
  const float*         action  = (const float*)d_in[4];
  const float*         emb     = (const float*)d_in[5];
  const float*         W_obst  = (const float*)d_in[6];
  const float*         b_obst  = (const float*)d_in[7];
  const float*         W_act   = (const float*)d_in[8];
  const float*         b_act   = (const float*)d_in[9];
  const float*         W_pos   = (const float*)d_in[10];
  const float*         b_pos   = (const float*)d_in[11];
  const float*         c1_Wl   = (const float*)d_in[12];
  const float*         c1_bl   = (const float*)d_in[13];
  const float*         c1_Wg   = (const float*)d_in[14];
  const float*         c1_bg   = (const float*)d_in[15];
  const float*         c2_Wl   = (const float*)d_in[16];
  const float*         c2_bl   = (const float*)d_in[17];
  const float*         c2_Wg   = (const float*)d_in[18];
  const float*         c2_bg   = (const float*)d_in[19];
  const float*         f_W1    = (const float*)d_in[20];
  const float*         f_b1    = (const float*)d_in[21];
  const float*         f_W2    = (const float*)d_in[22];
  const float*         f_b2    = (const float*)d_in[23];
  const float*         f_W3    = (const float*)d_in[24];
  const float*         f_b3    = (const float*)d_in[25];
  float*               outp    = (float*)d_out;
  half_t*              ws16    = (half_t*)d_ws;

  int B = in_sizes[0] / (NPRD * 2);   // 1024

  prep_weights<<<(6 * 16384) / 256, 256, 0, stream>>>(
      c1_Wl, c1_Wg, c2_Wl, c2_Wg, f_W1, f_W2, ws16);

  prey_critic_main<<<B, 128, 0, stream>>>(
      pred, prey, obst, alive, action,
      emb, W_obst, b_obst, W_act, b_act, W_pos, b_pos,
      c1_Wl, c1_bl, c1_bg, c2_Wl, c2_bl, c2_bg,
      f_b1, f_b2, f_W3, f_b3, ws16, outp);
}
